// PatentSubgraph_37993280700882
// MI455X (gfx1250) — matrix operand order repaired and checked by reading, not compiled, not run on hardware
//
#include <hip/hip_runtime.h>
#include <hip/hip_bf16.h>

typedef __attribute__((ext_vector_type(16))) _Float16 v16h;
typedef __attribute__((ext_vector_type(8)))  float    v8f;
typedef __attribute__((ext_vector_type(4)))  float    v4f;

#define DD 64
#define HH 64

// ---------------------------------------------------------------------------
// Address-space helpers: force global_load (not flat_load) for L2-resident
// tables & weight fragments so the LDS/flat pipe stays free for ds traffic.
// ---------------------------------------------------------------------------
__device__ __forceinline__ const __attribute__((address_space(1))) v4f*
gvec(const float* p) {
  return (const __attribute__((address_space(1))) v4f*)p;
}
__device__ __forceinline__ const __attribute__((address_space(1))) v16h*
gfrag(const _Float16* p) {
  return (const __attribute__((address_space(1))) v16h*)p;
}

// ---------------------------------------------------------------------------
// WMMA helpers (CDNA5 v_wmma_f32_16x16x32_f16, wave32)
// ---------------------------------------------------------------------------
__device__ __forceinline__ v8f wmma_f16(v16h a, v16h b, v8f c) {
  // (neg_a, A, neg_b, B, c_mod, C, reuse_a, reuse_b)
  return __builtin_amdgcn_wmma_f32_16x16x32_f16(false, a, false, b, (short)0, c,
                                                false, false);
}

// A-matrix fragment from an LDS [16][64] f16 tile (array-ref keeps addrspace(3)
// so ds_load_b128 is emitted with no generic-pointer null checks).
// ISA 16x32 f16 A layout: lane m=l&15; lanes0-15 K {0..7,16..23},
// lanes16-31 K {8..15,24..31}; element pairs per VGPR.
__device__ __forceinline__ v16h load_a_frag(const _Float16 (&tile)[16][DD],
                                            int lane, int kbase) {
  v16h a;
  int m  = lane & 15;
  int kh = (lane >> 4) << 3;
#pragma unroll
  for (int e = 0; e < 16; ++e) {
    int r = e >> 1, j = e & 1;
    int k = (r < 4) ? (kh + 2 * r + j) : (16 + kh + 2 * (r - 4) + j);
    a[e] = tile[m][kbase + k];
  }
  return a;
}

// B-matrix fragment: pre-swizzled, 32 lanes x 16 halves contiguous per
// fragment -> two aligned global_load_b128 per lane.
__device__ __forceinline__ v16h load_b_frag(const _Float16* base, int frag, int lane) {
  return *gfrag(base + ((size_t)frag * 32 + lane) * 16);
}

__device__ __forceinline__ float sigmoidf_(float x) {
  return 1.0f / (1.0f + __expf(-x));
}

// ---------------------------------------------------------------------------
// Weight pack: f32 (K x N) matrix -> f16 B-fragments, frag = kt*(N/16)+nt,
// element (k,n): k = kt*32 + (lane>>4)*8 + (e&7) + (e>>3)*16 ; n = nt*16 + (lane&15)
// trans=1 reads src as (N rows x K cols)   (for W_ih.T / W_hh.T)
// ---------------------------------------------------------------------------
__global__ void pack_b_frags(const float* __restrict__ src, _Float16* __restrict__ dst,
                             int K, int N, int trans) {
  int tid   = blockIdx.x * blockDim.x + threadIdx.x;
  int total = K * N;  // (K/32)*(N/16)*512
  if (tid >= total) return;
  int e    = tid & 15;
  int lane = (tid >> 4) & 31;
  int frag = tid >> 9;
  int NT   = N >> 4;
  int kt   = frag / NT, nt = frag - kt * NT;
  int k    = kt * 32 + ((lane >> 4) << 3) + (e & 7) + ((e >> 3) << 4);
  int n    = nt * 16 + (lane & 15);
  float v  = trans ? src[n * K + k] : src[k * N + n];
  dst[tid] = (_Float16)v;
}

// ---------------------------------------------------------------------------
// att_fuse on a 16-path tile held in LDS:
//   h = leaky_relu([e|sub] @ W1 + b1, 0.2); w = sigmoid(h@W2 + b2); e += sub*w
// ---------------------------------------------------------------------------
__device__ __forceinline__ void att_fuse(_Float16 (&et)[16][DD],
                                         const _Float16 (&st)[16][DD],
                                         const _Float16* fW1, const float* b1,
                                         const float* W2, float b2,
                                         float* wbuf, int lane) {
  v16h a0 = load_a_frag(et, lane, 0);
  v16h a1 = load_a_frag(et, lane, 32);
  v16h a2 = load_a_frag(st, lane, 0);
  v16h a3 = load_a_frag(st, lane, 32);
  int nloc = lane & 15;
  float hp[8];
#pragma unroll
  for (int r = 0; r < 8; ++r) hp[r] = 0.0f;
#pragma unroll
  for (int nt = 0; nt < 4; ++nt) {
    float bv = b1[nt * 16 + nloc];
    v8f acc;
#pragma unroll
    for (int r = 0; r < 8; ++r) acc[r] = bv;   // C-layout: n depends only on lane
    acc = wmma_f16(a0, load_b_frag(fW1, 0 * 4 + nt, lane), acc);
    acc = wmma_f16(a1, load_b_frag(fW1, 1 * 4 + nt, lane), acc);
    acc = wmma_f16(a2, load_b_frag(fW1, 2 * 4 + nt, lane), acc);
    acc = wmma_f16(a3, load_b_frag(fW1, 3 * 4 + nt, lane), acc);
    float w2v = W2[nt * 16 + nloc];
#pragma unroll
    for (int r = 0; r < 8; ++r) {
      float h = acc[r];
      h = (h > 0.0f) ? h : 0.2f * h;           // leaky_relu(0.2)
      hp[r] += h * w2v;
    }
  }
  // reduce the N-dimension across the 16 lanes of each half-wave
#pragma unroll
  for (int off = 8; off >= 1; off >>= 1)
#pragma unroll
    for (int r = 0; r < 8; ++r) hp[r] += __shfl_xor(hp[r], off, 32);
  if (nloc == 0) {
    int mb = (lane >> 4) << 3;
#pragma unroll
    for (int r = 0; r < 8; ++r) wbuf[mb + r] = sigmoidf_(hp[r] + b2);
  }
  __syncthreads();
  // e += sub * w : 2 lanes per path, 32 contiguous dims each
  {
    int p  = lane >> 1;
    int dh = (lane & 1) << 5;
    float w = wbuf[p];
#pragma unroll
    for (int q = 0; q < 32; ++q) {
      int d = dh + q;
      float e = (float)et[p][d];
      float s = (float)st[p][d];
      et[p][d] = (_Float16)(e + s * w);
    }
  }
  __syncthreads();
}

// ---------------------------------------------------------------------------
// Main fused kernel: one wave32 per block processes 16 paths end-to-end.
// ---------------------------------------------------------------------------
__global__ __launch_bounds__(32) void path_kernel(
    const int* __restrict__ paths,
    const int* __restrict__ pat_fp_idx,  const unsigned char* __restrict__ pat_fp_mask,
    const int* __restrict__ pat_ipc_idx, const unsigned char* __restrict__ pat_ipc_mask,
    const int* __restrict__ pat_comp_idx,const unsigned char* __restrict__ pat_comp_mask,
    const int* __restrict__ comp_ind_idx,const unsigned char* __restrict__ comp_ind_mask,
    const int* __restrict__ comp_pat_idx,const unsigned char* __restrict__ comp_pat_mask,
    const float* __restrict__ company_emb, const float* __restrict__ patent_emb,
    const float* __restrict__ fp_emb,      const float* __restrict__ ipc_emb,
    const float* __restrict__ ind_emb,
    const float* __restrict__ b_ih, const float* __restrict__ b_hh,
    const float* __restrict__ att_b1, const float* __restrict__ att_W2,
    const float* __restrict__ att_b2,
    const float* __restrict__ mlp_b1, const float* __restrict__ mlp_W2,
    const float* __restrict__ mlp_b2,
    const _Float16* __restrict__ fAtt, const _Float16* __restrict__ fIh,
    const _Float16* __restrict__ fHh,  const _Float16* __restrict__ fM1,
    float* __restrict__ out) {
  __shared__ _Float16 X[4][16][DD];     // sequence e0..e3 (f16)
  __shared__ _Float16 csub[16][DD];
  __shared__ _Float16 psub[16][DD];
  __shared__ _Float16 htile[16][HH];    // LSTM hidden restage
  __shared__ float    wbuf[16];

  int lane = threadIdx.x;
  int nloc = lane & 15;
  int base = blockIdx.x * 16;

  // ---- Stage 0: lazy gathers, 2 lanes per path x 32 contiguous dims -------
  // All rows loaded as 8x float4 (global_load_b128), tables are L2-resident.
  {
    int p  = lane >> 1;
    int dh = (lane & 1) << 5;
    int pi = base + p;
    int c  = paths[pi * 4 + 0];
    int p1 = paths[pi * 4 + 1];
    int f2 = paths[pi * 4 + 2];
    int p3 = paths[pi * 4 + 3];

    v4f e0[8], e3[8], aC[8], aP[8];
    {
      auto r = gvec(company_emb + (size_t)c * DD + dh);
#pragma unroll
      for (int q = 0; q < 8; ++q) e0[q] = r[q];
    }
    {
      auto r = gvec(patent_emb + (size_t)p1 * DD + dh);
#pragma unroll
      for (int q = 0; q < 8; ++q) {
        v4f x = r[q];
#pragma unroll
        for (int j = 0; j < 4; ++j) X[1][p][dh + q * 4 + j] = (_Float16)x[j];
      }
    }
    {
      auto r = gvec(fp_emb + (size_t)f2 * DD + dh);
#pragma unroll
      for (int q = 0; q < 8; ++q) {
        v4f x = r[q];
#pragma unroll
        for (int j = 0; j < 4; ++j) X[2][p][dh + q * 4 + j] = (_Float16)x[j];
      }
    }
    {
      auto r = gvec(patent_emb + (size_t)p3 * DD + dh);
#pragma unroll
      for (int q = 0; q < 8; ++q) e3[q] = r[q];
    }
#pragma unroll
    for (int q = 0; q < 8; ++q) {
#pragma unroll
      for (int j = 0; j < 4; ++j) {
        X[0][p][dh + q * 4 + j] = (_Float16)e0[q][j];
        X[3][p][dh + q * 4 + j] = (_Float16)e3[q][j];
      }
      aC[q] = (v4f){0.f, 0.f, 0.f, 0.f};
      aP[q] = (v4f){0.f, 0.f, 0.f, 0.f};
    }

    // company_sub[c]
    float cntC = 0.0f;
    for (int j = 0; j < 4; ++j)
      if (comp_ind_mask[c * 4 + j]) {
        auto r = gvec(ind_emb + (size_t)comp_ind_idx[c * 4 + j] * DD + dh);
#pragma unroll
        for (int q = 0; q < 8; ++q) aC[q] += r[q];
        cntC += 1.0f;
      }
    for (int j = 0; j < 16; ++j)
      if (comp_pat_mask[c * 16 + j]) {
        auto r = gvec(patent_emb + (size_t)comp_pat_idx[c * 16 + j] * DD + dh);
#pragma unroll
        for (int q = 0; q < 8; ++q) aC[q] += r[q];
        cntC += 1.0f;
      }
    float invC = 1.0f / cntC;
#pragma unroll
    for (int q = 0; q < 8; ++q)
#pragma unroll
      for (int j = 0; j < 4; ++j)
        csub[p][dh + q * 4 + j] = (_Float16)(e0[q][j] + aC[q][j] * invC);

    // patent_sub[p3]
    float cntP = 0.0f;
    for (int j = 0; j < 2; ++j)
      if (pat_fp_mask[p3 * 2 + j]) {
        auto r = gvec(fp_emb + (size_t)pat_fp_idx[p3 * 2 + j] * DD + dh);
#pragma unroll
        for (int q = 0; q < 8; ++q) aP[q] += r[q];
        cntP += 1.0f;
      }
    for (int j = 0; j < 6; ++j)
      if (pat_ipc_mask[p3 * 6 + j]) {
        auto r = gvec(ipc_emb + (size_t)pat_ipc_idx[p3 * 6 + j] * DD + dh);
#pragma unroll
        for (int q = 0; q < 8; ++q) aP[q] += r[q];
        cntP += 1.0f;
      }
    for (int j = 0; j < 2; ++j)
      if (pat_comp_mask[p3 * 2 + j]) {
        auto r = gvec(company_emb + (size_t)pat_comp_idx[p3 * 2 + j] * DD + dh);
#pragma unroll
        for (int q = 0; q < 8; ++q) aP[q] += r[q];
        cntP += 1.0f;
      }
    float invP = 1.0f / cntP;
#pragma unroll
    for (int q = 0; q < 8; ++q)
#pragma unroll
      for (int j = 0; j < 4; ++j)
        psub[p][dh + q * 4 + j] = (_Float16)(e3[q][j] + aP[q][j] * invP);
  }
  __syncthreads();

  // ---- Stage 1: attention fusion (4x WMMA-tiled [16x128]@[128x64]) ----
  float ab2 = att_b2[0];
  att_fuse(X[1], csub, fAtt, att_b1, att_W2, ab2, wbuf, lane);
  att_fuse(X[1], psub, fAtt, att_b1, att_W2, ab2, wbuf, lane);
  att_fuse(X[2], csub, fAtt, att_b1, att_W2, ab2, wbuf, lane);
  att_fuse(X[2], psub, fAtt, att_b1, att_W2, ab2, wbuf, lane);

  // ---- Stage 2: 4-step LSTM, gate math stays in C-fragment registers ----
  // Gate bias is t-invariant in C-layout (N depends only on lane): hoist it.
  float bsum[16];
#pragma unroll
  for (int gate = 0; gate < 4; ++gate)
#pragma unroll
    for (int nt = 0; nt < 4; ++nt) {
      int n = gate * 64 + nt * 16 + nloc;
      bsum[gate * 4 + nt] = b_ih[n] + b_hh[n];
    }

  v8f cfr[4];
#pragma unroll
  for (int nt = 0; nt < 4; ++nt)
#pragma unroll
    for (int r = 0; r < 8; ++r) cfr[nt][r] = 0.0f;

#pragma unroll
  for (int t = 0; t < 4; ++t) {
    v16h ax0 = load_a_frag(X[t], lane, 0);
    v16h ax1 = load_a_frag(X[t], lane, 32);
    v16h ah0 = {};
    v16h ah1 = {};
    if (t > 0) {
      ah0 = load_a_frag(htile, lane, 0);
      ah1 = load_a_frag(htile, lane, 32);
    }
    __syncthreads();
#pragma unroll
    for (int nt = 0; nt < 4; ++nt) {
      v8f g4[4];
#pragma unroll
      for (int gate = 0; gate < 4; ++gate) {
        float bv = bsum[gate * 4 + nt];
        v8f acc;
#pragma unroll
        for (int r = 0; r < 8; ++r) acc[r] = bv;
        acc = wmma_f16(ax0, load_b_frag(fIh, gate * 4 + nt, lane), acc);
        acc = wmma_f16(ax1, load_b_frag(fIh, 16 + gate * 4 + nt, lane), acc);
        if (t > 0) {
          acc = wmma_f16(ah0, load_b_frag(fHh, gate * 4 + nt, lane), acc);
          acc = wmma_f16(ah1, load_b_frag(fHh, 16 + gate * 4 + nt, lane), acc);
        }
        g4[gate] = acc;
      }
      // i,f,g,o tiles share (lane, vgpr-slot) coordinates -> register math
#pragma unroll
      for (int r = 0; r < 8; ++r) {
        float iv = sigmoidf_(g4[0][r]);
        float fv = sigmoidf_(g4[1][r]);
        float gv = tanhf(g4[2][r]);
        float ov = sigmoidf_(g4[3][r]);
        float cn = fv * cfr[nt][r] + iv * gv;
        cfr[nt][r] = cn;
        float hn = ov * tanhf(cn);
        int m = r + ((lane >> 4) << 3);
        htile[m][nt * 16 + nloc] = (_Float16)hn;  // restage h as next A operand
      }
    }
    __syncthreads();
  }

  // ---- Stage 3: MLP head: relu(h@W1+b1)@W2+b2 -> sigmoid ----
  v16h ah0 = load_a_frag(htile, lane, 0);
  v16h ah1 = load_a_frag(htile, lane, 32);
  float sp[8];
#pragma unroll
  for (int r = 0; r < 8; ++r) sp[r] = 0.0f;
#pragma unroll
  for (int nt = 0; nt < 4; ++nt) {
    float bv = mlp_b1[nt * 16 + nloc];
    v8f acc;
#pragma unroll
    for (int r = 0; r < 8; ++r) acc[r] = bv;
    acc = wmma_f16(ah0, load_b_frag(fM1, 0 * 4 + nt, lane), acc);
    acc = wmma_f16(ah1, load_b_frag(fM1, 1 * 4 + nt, lane), acc);
    float w2v = mlp_W2[nt * 16 + nloc];
#pragma unroll
    for (int r = 0; r < 8; ++r) sp[r] += fmaxf(acc[r], 0.0f) * w2v;
  }
#pragma unroll
  for (int off = 8; off >= 1; off >>= 1)
#pragma unroll
    for (int r = 0; r < 8; ++r) sp[r] += __shfl_xor(sp[r], off, 32);
  if (nloc == 0) {
    int mb = (lane >> 4) << 3;
    float b2 = mlp_b2[0];
#pragma unroll
    for (int r = 0; r < 8; ++r) out[base + mb + r] = sigmoidf_(sp[r] + b2);
  }
}

// ---------------------------------------------------------------------------
extern "C" void kernel_launch(void* const* d_in, const int* in_sizes, int n_in,
                              void* d_out, int out_size, void* d_ws, size_t ws_size,
                              hipStream_t stream) {
  (void)n_in; (void)out_size; (void)ws_size;
  const int P = in_sizes[0] / 4;

  // workspace layout (f16 halves): pre-swizzled B fragments
  _Float16* ws   = (_Float16*)d_ws;
  _Float16* fAtt = ws;          // att_W1  (K=128,N=64)  : 8192
  _Float16* fIh  = ws + 8192;   // W_ih.T  (K=64, N=256) : 16384
  _Float16* fHh  = ws + 24576;  // W_hh.T  (K=64, N=256) : 16384
  _Float16* fM1  = ws + 40960;  // mlp_W1  (K=64, N=64)  : 4096

  pack_b_frags<<<(128 * 64 + 255) / 256, 256, 0, stream>>>(
      (const float*)d_in[21], fAtt, 128, 64, 0);
  pack_b_frags<<<(64 * 256 + 255) / 256, 256, 0, stream>>>(
      (const float*)d_in[17], fIh, 64, 256, 1);
  pack_b_frags<<<(64 * 256 + 255) / 256, 256, 0, stream>>>(
      (const float*)d_in[18], fHh, 64, 256, 1);
  pack_b_frags<<<(64 * 64 + 255) / 256, 256, 0, stream>>>(
      (const float*)d_in[25], fM1, 64, 64, 0);

  path_kernel<<<P / 16, 32, 0, stream>>>(
      (const int*)d_in[0],
      (const int*)d_in[2],  (const unsigned char*)d_in[3],
      (const int*)d_in[4],  (const unsigned char*)d_in[5],
      (const int*)d_in[6],  (const unsigned char*)d_in[7],
      (const int*)d_in[8],  (const unsigned char*)d_in[9],
      (const int*)d_in[10], (const unsigned char*)d_in[11],
      (const float*)d_in[12], (const float*)d_in[13], (const float*)d_in[14],
      (const float*)d_in[15], (const float*)d_in[16],
      (const float*)d_in[19], (const float*)d_in[20],
      (const float*)d_in[22], (const float*)d_in[23], (const float*)d_in[24],
      (const float*)d_in[26], (const float*)d_in[27], (const float*)d_in[28],
      fAtt, fIh, fHh, fM1,
      (float*)d_out);
}
